// DerivedModel_26242250179117
// MI455X (gfx1250) — matrix operand order repaired and checked by reading, not compiled, not run on hardware
//
#include <hip/hip_runtime.h>
#include <hip/hip_bf16.h>
#include <cstdint>
#include <cstddef>

typedef __bf16 bf16_t;
typedef __bf16 bf16x8  __attribute__((ext_vector_type(8)));
typedef __bf16 bf16x16 __attribute__((ext_vector_type(16)));
typedef float  f32x8   __attribute__((ext_vector_type(8)));

#define NB 8
#define NT 12
#define NH 64
#define NW 64
#define NF 64
#define C4 256
#define HW (NH*NW)     // 4096
#define PIX (NB*HW)    // 32768
#define PC 72          // padded channel stride in LDS patches (bank spread)

static __device__ __forceinline__ bf16x16 cat8(bf16x8 lo, bf16x8 hi) {
  return __builtin_shufflevector(lo, hi, 0,1,2,3,4,5,6,7,8,9,10,11,12,13,14,15);
}

// ---------------- f32 -> bf16 conversion ----------------
__global__ void cvt_bf16_kernel(const float* __restrict__ in, bf16_t* __restrict__ out, int n) {
  int i = blockIdx.x * blockDim.x + threadIdx.x;
  if (i < n) out[i] = (bf16_t)in[i];
}

// ---------------- zero h (bf16) and c (f32) ----------------
__global__ void zero_state_kernel(bf16_t* __restrict__ h, float* __restrict__ c, int n) {
  int i = blockIdx.x * blockDim.x + threadIdx.x;
  if (i < n) { h[i] = (bf16_t)0.f; c[i] = 0.f; }
}

// ---------------- repack [3][3][Cx][256] + [3][3][64][256] f32 -> Wt[256][Ktot] bf16 ----------------
__global__ void repack_w_kernel(const float* __restrict__ kw, const float* __restrict__ rkw,
                                bf16_t* __restrict__ Wt, int Cx, int lgCx, int KpadX, int Ktot) {
  int co = blockIdx.x;   // 0..255
  for (int k = threadIdx.x; k < Ktot; k += blockDim.x) {
    float v = 0.f;
    if (k < KpadX) {
      if (k < (9 << lgCx)) {
        int c = k & (Cx - 1), w9 = k >> lgCx;
        v = kw[(w9 * Cx + c) * C4 + co];
      }
    } else {
      int kk = k - KpadX;
      int c = kk & 63, w9 = kk >> 6;
      v = rkw[(w9 * 64 + c) * C4 + co];
    }
    Wt[(size_t)co * Ktot + k] = (bf16_t)v;
  }
}

// One K-chunk (32 K) of the implicit GEMM: A from LDS patch at a compile-time
// tap/channel offset, B from global at a compile-time byte offset, 8 WMMAs.
static __device__ __forceinline__ void gemm_chunk(
    const bf16_t* __restrict__ P,      // LDS patch (Xp or Hp)
    int sec_cc,                        // chunk index within section (compile-time)
    int gch,                           // global chunk index (compile-time) -> B offset
    int a0base, int a1base,            // per-lane LDS element bases (lm*PC+ak0, (lm+16)*PC+ak0)
    const bf16_t* const (&brow)[4],    // per-nt B row base pointers (+bk0)
    f32x8 (&acc)[2][4])
{
  const int tap   = sec_cc >> 1;
  const int c_off = (sec_cc & 1) << 5;
  const int ky = tap / 3, kx = tap - 3 * ky;
  const int u  = (ky * 34 + kx) * PC + c_off;      // constant after unroll
  bf16x8 l0 = *(const bf16x8*)&P[a0base + u];
  bf16x8 h0 = *(const bf16x8*)&P[a0base + u + 16];
  bf16x8 l1 = *(const bf16x8*)&P[a1base + u];
  bf16x8 h1 = *(const bf16x8*)&P[a1base + u + 16];
  bf16x16 a0 = cat8(l0, h0);
  bf16x16 a1 = cat8(l1, h1);
  #pragma unroll
  for (int nt = 0; nt < 4; ++nt) {
    const bf16_t* bp = brow[nt] + gch * 32;        // folds to immediate offset
    bf16x8 blo = *(const bf16x8*)(bp);
    bf16x8 bhi = *(const bf16x8*)(bp + 16);
    bf16x16 bm = cat8(blo, bhi);
    acc[0][nt] = __builtin_amdgcn_wmma_f32_16x16x32_bf16(
        false, a0, false, bm, (short)0, acc[0][nt], false, false);
    acc[1][nt] = __builtin_amdgcn_wmma_f32_16x16x32_bf16(
        false, a1, false, bm, (short)0, acc[1][nt], false, false);
  }
}

// ---------------- fused input+recurrent implicit-GEMM (WMMA bf16) ----------------
// Block: 128 threads (4 waves), 32 pixels (2 M-tiles), all 256 gate channels.
// Input patches (3 x 34 x C) for x and h loaded to LDS once; K-loop fully
// unrolled at compile time (CX/KTOT template params) -> immediate-offset loads.
template<int CX, int KTOT>
__global__ __launch_bounds__(128) void convlstm_gemm_kernel(
    const bf16_t* __restrict__ xin, long xin_bstride,
    const bf16_t* __restrict__ hin,
    const bf16_t* __restrict__ Wt,
    const float* __restrict__ bias,
    float* __restrict__ z)
{
  __shared__ __align__(16) bf16_t Xp[3 * 34 * PC];   // x patch (CX==64 path)
  __shared__ __align__(16) bf16_t Hp[3 * 34 * PC];   // h patch
  __shared__ __align__(16) bf16_t As2[32 * 40];      // layer-1 x-section staged tile

  const int tid  = threadIdx.x;
  const int wave = tid >> 5;
  const int lane = tid & 31;

  const int p0 = blockIdx.x << 5;          // 32 consecutive pixels along W
  const int b  = p0 >> 12;
  const int y  = (p0 & 4095) >> 6;
  const int x0 = p0 & 63;                  // 0 or 32

  const long hin_b = (long)b * (long)(HW * NF);
  const long xin_b = (long)b * xin_bstride;

  const bf16_t bz = (bf16_t)0.f;

  // ---- one-time patch fill: rows y-1..y+1, cols x0-1..x0+32, vec8 channels ----
  for (int g = tid; g < 3 * 34 * 8; g += 128) {
    int r  = g / 272;
    int t2 = g - r * 272;
    int px = t2 >> 3;
    int cg = (t2 & 7) << 3;
    int iy = y + r - 1;
    int ix = x0 + px - 1;
    bool inb = ((unsigned)iy < (unsigned)NH) && ((unsigned)ix < (unsigned)NW);
    bf16x8 vh = {bz,bz,bz,bz,bz,bz,bz,bz};
    if (inb) vh = *(const bf16x8*)&hin[hin_b + (((long)iy * NW + ix) << 6) + cg];
    *(bf16x8*)&Hp[(r * 34 + px) * PC + cg] = vh;
    if (CX == 64) {
      bf16x8 vx = {bz,bz,bz,bz,bz,bz,bz,bz};
      if (inb) vx = *(const bf16x8*)&xin[xin_b + (((long)iy * NW + ix) << 6) + cg];
      *(bf16x8*)&Xp[(r * 34 + px) * PC + cg] = vx;
    }
  }
  if (CX == 1) {  // layer-1 x-section: 32 pixels x 32 K (9 real taps, rest zero)
    for (int idx = tid; idx < 32 * 32; idx += 128) {
      int j = idx >> 5, k = idx & 31;
      bf16_t v = bz;
      if (k < 9) {
        int ky = k / 3, kx = k - 3 * ky;
        int iy = y + ky - 1, ix = x0 + j + kx - 1;
        if ((unsigned)iy < (unsigned)NH && (unsigned)ix < (unsigned)NW)
          v = xin[xin_b + (long)iy * NW + ix];
      }
      As2[j * 40 + k] = v;
    }
  }
  __syncthreads();

  const int lm  = lane & 15;
  const int ak0 = (lane >> 4) << 3;     // A K-offset per ISA 16-bit A layout
  const int bk0 = (lane >> 4) << 3;     // B K-offset (same split layout as A)

  const int a0base = lm * PC + ak0;
  const int a1base = (lm + 16) * PC + ak0;

  f32x8 acc[2][4];
  #pragma unroll
  for (int mt = 0; mt < 2; ++mt)
    #pragma unroll
    for (int nt = 0; nt < 4; ++nt)
      acc[mt][nt] = f32x8{0.f,0.f,0.f,0.f,0.f,0.f,0.f,0.f};

  // B row base pointers (constant per kernel; chunk offsets are immediates)
  const bf16_t* brow[4];
  #pragma unroll
  for (int nt = 0; nt < 4; ++nt)
    brow[nt] = Wt + (size_t)((wave << 6) + (nt << 4) + lm) * KTOT + bk0;

  int gch = 0;
  if (CX == 1) {
    // special chunk 0 from As2 (layer-1 x-section)
    bf16x8 l0 = *(const bf16x8*)&As2[lm * 40 + ak0];
    bf16x8 h0 = *(const bf16x8*)&As2[lm * 40 + ak0 + 16];
    bf16x8 l1 = *(const bf16x8*)&As2[(16 + lm) * 40 + ak0];
    bf16x8 h1 = *(const bf16x8*)&As2[(16 + lm) * 40 + ak0 + 16];
    bf16x16 a0 = cat8(l0, h0);
    bf16x16 a1 = cat8(l1, h1);
    #pragma unroll
    for (int nt = 0; nt < 4; ++nt) {
      bf16x8 blo = *(const bf16x8*)(brow[nt]);
      bf16x8 bhi = *(const bf16x8*)(brow[nt] + 16);
      bf16x16 bm = cat8(blo, bhi);
      acc[0][nt] = __builtin_amdgcn_wmma_f32_16x16x32_bf16(
          false, a0, false, bm, (short)0, acc[0][nt], false, false);
      acc[1][nt] = __builtin_amdgcn_wmma_f32_16x16x32_bf16(
          false, a1, false, bm, (short)0, acc[1][nt], false, false);
    }
    gch = 1;
  } else {
    // x-section: 18 chunks from Xp
    #pragma unroll
    for (int cc = 0; cc < 18; ++cc)
      gemm_chunk(Xp, cc, cc, a0base, a1base, brow, acc);
    gch = 18;
  }
  // h-section: 18 chunks from Hp
  #pragma unroll
  for (int cc = 0; cc < 18; ++cc)
    gemm_chunk(Hp, cc, gch + cc, a0base, a1base, brow, acc);

  // ---- epilogue: C layout VGPR r -> M = r + 8*(lane>=16), N = lane&15 ----
  const int n  = lane & 15;
  const int mb = (lane >> 4) << 3;
  #pragma unroll
  for (int mt = 0; mt < 2; ++mt) {
    #pragma unroll
    for (int nt = 0; nt < 4; ++nt) {
      int co = (wave << 6) + (nt << 4) + n;
      float bv = bias[co];
      #pragma unroll
      for (int r = 0; r < 8; ++r) {
        int pix = p0 + (mt << 4) + mb + r;
        z[(size_t)pix * C4 + co] = acc[mt][nt][r] + bv;
      }
    }
  }
}

// ---------------- LSTM gates + BN(frozen)+ReLU epilogue ----------------
__global__ void gates_kernel(const float* __restrict__ z, float* __restrict__ c,
                             bf16_t* __restrict__ h,
                             const float* __restrict__ gamma, const float* __restrict__ beta,
                             bf16_t* __restrict__ seq_out, int t)
{
  int idx = blockIdx.x * blockDim.x + threadIdx.x;  // p*64 + f
  if (idx >= PIX * NF) return;
  int f = idx & 63;
  int p = idx >> 6;
  const float* zp = z + (size_t)p * C4;
  float zi = zp[f], zf = zp[64 + f], zg = zp[128 + f], zo = zp[192 + f];
  float si = 1.f / (1.f + __expf(-zi));
  float sf = 1.f / (1.f + __expf(-zf));
  float so = 1.f / (1.f + __expf(-zo));
  float cn = sf * c[idx] + si * tanhf(zg);
  float hn = so * tanhf(cn);
  c[idx] = cn;
  h[idx] = (bf16_t)hn;
  float o = gamma[f] * hn * 0.99950037f + beta[f];   // 1/sqrt(1+1e-3)
  o = o > 0.f ? o : 0.f;
  int b = p >> 12;
  size_t so_idx = (((size_t)(b * NT + t)) * HW + (size_t)(p & 4095)) * NF + f;
  seq_out[so_idx] = (bf16_t)o;
}

// ---------------- conv3d 3x3x3 (Cout=1) + sigmoid ----------------
__global__ void conv3d_sig_kernel(const bf16_t* __restrict__ seq, const float* __restrict__ w3,
                                  const float* __restrict__ b3, float* __restrict__ out)
{
  int idx = blockIdx.x * blockDim.x + threadIdx.x;  // ((b*T + t)*H + y)*W + x
  if (idx >= NB * NT * HW) return;
  int x = idx & 63;
  int y = (idx >> 6) & 63;
  int t = (idx >> 12) % NT;
  int b = idx / (NT * HW);
  float acc = b3[0];
  for (int td = 0; td < 3; ++td) {
    int tt = t + td - 1; if ((unsigned)tt >= (unsigned)NT) continue;
    for (int ky = 0; ky < 3; ++ky) {
      int yy = y + ky - 1; if ((unsigned)yy >= (unsigned)NH) continue;
      for (int kx = 0; kx < 3; ++kx) {
        int xx = x + kx - 1; if ((unsigned)xx >= (unsigned)NW) continue;
        const bf16_t* sp = seq + (((size_t)(b * NT + tt) * HW + (size_t)yy * NW + xx) << 6);
        const float*  wp = w3 + ((td * 3 + ky) * 3 + kx) * 64;
        #pragma unroll 8
        for (int cc = 0; cc < 64; ++cc) acc += wp[cc] * (float)sp[cc];
      }
    }
  }
  out[idx] = 1.f / (1.f + __expf(-acc));
}

// ---------------- host launch ----------------
extern "C" void kernel_launch(void* const* d_in, const int* in_sizes, int n_in,
                              void* d_out, int out_size, void* d_ws, size_t ws_size,
                              hipStream_t stream)
{
  (void)in_sizes; (void)n_in; (void)out_size; (void)ws_size;
  const float* x     = (const float*)d_in[0];
  const float* kw[3] = {(const float*)d_in[1], (const float*)d_in[6],  (const float*)d_in[11]};
  const float* rk[3] = {(const float*)d_in[2], (const float*)d_in[7],  (const float*)d_in[12]};
  const float* bb[3] = {(const float*)d_in[3], (const float*)d_in[8],  (const float*)d_in[13]};
  const float* gg[3] = {(const float*)d_in[4], (const float*)d_in[9],  (const float*)d_in[14]};
  const float* be[3] = {(const float*)d_in[5], (const float*)d_in[10], (const float*)d_in[15]};
  const float* w3 = (const float*)d_in[16];
  const float* b3 = (const float*)d_in[17];
  float* out = (float*)d_out;

  char* wsp = (char*)d_ws;
  size_t off = 0;
  auto carve = [&](size_t bytes) -> char* {
    char* p = wsp + off;
    off = (off + bytes + 255) & ~(size_t)255;
    return p;
  };
  bf16_t* xbf  = (bf16_t*)carve((size_t)NB * NT * HW * 2);            // x as bf16 (Cin=1)
  bf16_t* seqA = (bf16_t*)carve((size_t)NB * NT * HW * NF * 2);       // ping
  bf16_t* seqB = (bf16_t*)carve((size_t)NB * NT * HW * NF * 2);       // pong
  float*  z    = (float*) carve((size_t)PIX * C4 * 4);                // gate pre-activations
  bf16_t* h    = (bf16_t*)carve((size_t)PIX * NF * 2);                // hidden state (bf16)
  float*  c    = (float*) carve((size_t)PIX * NF * 4);                // cell state
  bf16_t* Wt   = (bf16_t*)carve((size_t)C4 * 1152 * 2);               // packed weights

  { int n = NB * NT * HW; cvt_bf16_kernel<<<(n + 255) / 256, 256, 0, stream>>>(x, xbf, n); }

  for (int l = 0; l < 3; ++l) {
    int Cx    = (l == 0) ? 1 : 64;
    int lgCx  = (l == 0) ? 0 : 6;
    int KpadX = (l == 0) ? 32 : 576;   // round_up(9*Cx, 32)
    int Ktot  = KpadX + 576;           // + 9*64 recurrent section

    repack_w_kernel<<<C4, 256, 0, stream>>>(kw[l], rk[l], Wt, Cx, lgCx, KpadX, Ktot);
    { int n = PIX * NF; zero_state_kernel<<<(n + 255) / 256, 256, 0, stream>>>(h, c, n); }

    const bf16_t* xin_base = (l == 0) ? xbf : ((l == 1) ? seqA : seqB);
    bf16_t* seq_out = (l == 1) ? seqB : seqA;      // l0->A, l1->B, l2->A
    long tstride = (long)HW * Cx;
    long bstride = (long)NT * HW * Cx;

    for (int t = 0; t < NT; ++t) {
      if (l == 0) {
        convlstm_gemm_kernel<1, 608><<<PIX / 32, 128, 0, stream>>>(
            xin_base + (size_t)t * tstride, bstride, h, Wt, bb[l], z);
      } else {
        convlstm_gemm_kernel<64, 1152><<<PIX / 32, 128, 0, stream>>>(
            xin_base + (size_t)t * tstride, bstride, h, Wt, bb[l], z);
      }
      int n = PIX * NF;
      gates_kernel<<<(n + 255) / 256, 256, 0, stream>>>(z, c, h, gg[l], be[l], seq_out, t);
    }
  }

  { int n = NB * NT * HW; conv3d_sig_kernel<<<(n + 255) / 256, 256, 0, stream>>>(seqA, w3, b3, out); }
}
